// SharedWorkspace_19636590477498
// MI455X (gfx1250) — compile-verified, matrix operand order checked by reading, not run on hardware
//
#include <hip/hip_runtime.h>
#include <hip/hip_bf16.h>

// ---------------- problem constants ----------------
#define NB   4096
#define KM   16
#define KH   64
#define DD   256
#define HH_N 4
#define DKV  64

// ---------------- LDS layout strides (padded, bank-conflict-free, 16B aligned)
#define LDW  264   // rows of [.,256] bf16 buffers
#define LDP  72    // write-probs / wvT rows (64 + pad)
#define LDR  40    // read-probs / rvT rows (32 + pad)
#define LDSW 72    // write-score fp32 rows
#define LDSR 24    // read-score fp32 rows

typedef __bf16 bf16_t;
typedef bf16_t v16bf __attribute__((ext_vector_type(16)));
typedef bf16_t v8bf  __attribute__((ext_vector_type(8)));
typedef float  v8f   __attribute__((ext_vector_type(8)));

// ---- LDS/global operand tile load: 16x32 bf16 (A layout; B uses same per-lane
// layout with W pre-transposed to N-major). lane<16: K kb..kb+7 & kb+16..kb+23.
__device__ __forceinline__ v16bf ld_tile(const bf16_t* p, int lda) {
  const int lane = threadIdx.x & 31;
  const int r  = lane & 15;
  const int kb = (lane >> 4) << 3;
  const bf16_t* q = p + r * lda + kb;
  v8bf lo = *(const v8bf*)(q);
  v8bf hi = *(const v8bf*)(q + 16);
  return __builtin_shufflevector(lo, hi, 0,1,2,3,4,5,6,7,8,9,10,11,12,13,14,15);
}

// ---- tile load through the "no-transpose view" head mapping:
// logical row t -> LDS row t>>2, column base (t&3)*64 (buffers with ld=LDW)
__device__ __forceinline__ v16bf ld_tile_map(const bf16_t* buf, int t0, int k0) {
  const int lane = threadIdx.x & 31;
  const int t  = t0 + (lane & 15);
  const int kb = ((lane >> 4) << 3) + k0;
  const bf16_t* q = buf + (t >> 2) * LDW + ((t & 3) << 6) + kb;
  v8bf lo = *(const v8bf*)(q);
  v8bf hi = *(const v8bf*)(q + 16);
  return __builtin_shufflevector(lo, hi, 0,1,2,3,4,5,6,7,8,9,10,11,12,13,14,15);
}

// ---- C/D 16x16 f32 accumulator store: lane<16 -> M 0..7, lane>=16 -> M 8..15
__device__ __forceinline__ void st_tile_bf(bf16_t* p, int ldo, v8f c) {
  const int lane = threadIdx.x & 31;
  const int nn = lane & 15;
  const int mb = (lane >> 4) << 3;
#pragma unroll
  for (int j = 0; j < 8; ++j) p[(mb + j) * ldo + nn] = (bf16_t)c[j];
}
__device__ __forceinline__ void st_tile_f32(float* p, int ldo, v8f c) {
  const int lane = threadIdx.x & 31;
  const int nn = lane & 15;
  const int mb = (lane >> 4) << 3;
#pragma unroll
  for (int j = 0; j < 8; ++j) p[(mb + j) * ldo + nn] = c[j];
}

template<int KS>
__device__ __forceinline__ v8f mm_tile(const bf16_t* Ab, int lda,
                                       const bf16_t* Bb, int ldb) {
  v8f acc = {};
#pragma unroll
  for (int ks = 0; ks < KS; ++ks) {
    v16bf a = ld_tile(Ab + ks * 32, lda);
    v16bf b = ld_tile(Bb + ks * 32, ldb);
    acc = __builtin_amdgcn_wmma_f32_16x16x32_bf16(false, a, false, b,
                                                  (short)0, acc, false, false);
  }
  return acc;
}

// -------- LDS region sizes (elements) --------
constexpr int SH_E  = 64 * LDW;          // h bf16
constexpr int SB_E  = 64 * LDW;          // wk -> rq
constexpr int SU_E  = HH_N * 64 * LDP;   // wvT  (>= 64*LDW for h_upd reuse)
constexpr int SD_E  = 16 * LDW;          // wq -> rk
constexpr int SE_E  = 16 * LDW;          // mem_upd
constexpr int SG_E  = 16 * LDW;          // M bf16 -> M_new bf16
constexpr int SS_F  = HH_N * 64 * LDSR;  // scores fp32 (>= 4*16*LDSW)
constexpr int SP_E  = HH_N * 64 * LDR;   // write probs -> rvT
constexpr int SPR_E = HH_N * 64 * LDR;   // read probs (K padded to 32)
constexpr size_t SMEM_BYTES =
    (size_t)(SH_E + SB_E + SU_E + SD_E + SE_E + SG_E + SP_E + SPR_E) * 2 +
    (size_t)SS_F * 4;

// -------- weight fp32 -> bf16 transposed (N-major) --------
__global__ __launch_bounds__(256) void wcvt_kernel(const float* __restrict__ src,
                                                   bf16_t* __restrict__ dst) {
  int e = blockIdx.x * 256 + threadIdx.x;   // 65536 elements
  int k = e >> 8, nn = e & 255;
  dst[nn * 256 + k] = (bf16_t)src[e];
}

__global__ __launch_bounds__(256)
void ws_attn_kernel(const float* __restrict__ Mg, const float* __restrict__ hg,
                    const float* __restrict__ maskg,
                    const bf16_t* __restrict__ Wt,
                    float* __restrict__ outM, float* __restrict__ outH) {
  extern __shared__ char smem_raw[];
  const int n   = blockIdx.x;
  const int tid = threadIdx.x;
  const int wid = tid >> 5;
  const int lane = tid & 31;

  bf16_t* sH  = (bf16_t*)smem_raw;
  bf16_t* sB  = sH + SH_E;
  bf16_t* sU  = sB + SB_E;          // wvT, later h_upd[64][LDW]
  bf16_t* sD  = sU + SU_E;
  bf16_t* sE  = sD + SD_E;
  bf16_t* sG  = sE + SE_E;
  float*  sS  = (float*)(sG + SG_E);
  bf16_t* sP  = (bf16_t*)(sS + SS_F);   // write probs, later rvT
  bf16_t* sPr = sP + SP_E;

  const bf16_t* WwkT = Wt + 0 * 65536;
  const bf16_t* WwvT = Wt + 1 * 65536;
  const bf16_t* WwqT = Wt + 2 * 65536;
  const bf16_t* WwoT = Wt + 3 * 65536;
  const bf16_t* WrkT = Wt + 4 * 65536;
  const bf16_t* WrvT = Wt + 5 * 65536;
  const bf16_t* WrqT = Wt + 6 * 65536;
  const bf16_t* WroT = Wt + 7 * 65536;

  const float* Mn = Mg + (size_t)n * KM * DD;
  const float* hn = hg + (size_t)n * KH * DD;
  const float* mk = maskg + (size_t)n * KH;
  float* oM = outM + (size_t)n * KM * DD;
  float* oH = outH + (size_t)n * KH * DD;

  // ---- stage 0: inputs -> bf16 LDS ----
  for (int e = tid; e < KH * DD; e += 256)
    sH[(e >> 8) * LDW + (e & 255)] = (bf16_t)hn[e];
  for (int e = tid; e < KM * DD; e += 256)
    sG[(e >> 8) * LDW + (e & 255)] = (bf16_t)Mn[e];
  __syncthreads();

  // ---- write-phase projections ----
  for (int t = wid; t < 64; t += 8) {              // wk = h @ Wwk -> sB
    int m0 = t >> 4, n0 = t & 15;
    v8f acc = mm_tile<8>(sH + m0 * 16 * LDW, LDW, WwkT + n0 * 16 * DD, DD);
    st_tile_bf(sB + m0 * 16 * LDW + n0 * 16, LDW, acc);
  }
  for (int t = wid; t < 64; t += 8) {              // wv -> wvT scatter
    int m0 = t >> 4, n0 = t & 15;
    v8f acc = mm_tile<8>(sH + m0 * 16 * LDW, LDW, WwvT + n0 * 16 * DD, DD);
    int nn = lane & 15, mb = (lane >> 4) << 3;
    int col = n0 * 16 + nn;
#pragma unroll
    for (int j = 0; j < 8; ++j) {
      int row = m0 * 16 + mb + j;
      int hh = row >> 4;
      int k  = ((row & 15) << 2) + (col >> 6);
      int v  = col & 63;
      sU[hh * (64 * LDP) + v * LDP + k] = (bf16_t)acc[j];
    }
  }
  for (int t = wid; t < 16; t += 8) {              // wq = M @ Wwq -> sD
    v8f acc = mm_tile<8>(sG, LDW, WwqT + t * 16 * DD, DD);
    st_tile_bf(sD + t * 16, LDW, acc);
  }
  __syncthreads();

  // ---- write scores: [16q x 64k] per head ----
  for (int t = wid; t < 16; t += 8) {
    int hh = t >> 2, n0 = t & 3;
    v8f acc = {};
#pragma unroll
    for (int ks = 0; ks < 2; ++ks) {
      v16bf a = ld_tile_map(sD, hh * 16, ks * 32);
      v16bf b = ld_tile_map(sB, hh * 64 + n0 * 16, ks * 32);
      acc = __builtin_amdgcn_wmma_f32_16x16x32_bf16(false, a, false, b,
                                                    (short)0, acc, false, false);
    }
    st_tile_f32(sS + hh * (16 * LDSW) + n0 * 16, LDSW, acc);
  }
  __syncthreads();

  // ---- write softmax: softmax((score/8) * mask) over 64 keys ----
  {
    int row = tid >> 2, part = tid & 3;
    int hh = row >> 4, q = row & 15;
    const float* srow = sS + hh * (16 * LDSW) + q * LDSW;
    float vals[16];
    float mx = -3.4e38f;
#pragma unroll
    for (int j = 0; j < 16; ++j) {
      int k = part * 16 + j;
      float s = srow[k] * 0.125f * mk[k];
      vals[j] = s;
      mx = fmaxf(mx, s);
    }
    mx = fmaxf(mx, __shfl_xor(mx, 1, 32));
    mx = fmaxf(mx, __shfl_xor(mx, 2, 32));
    float sum = 0.f;
#pragma unroll
    for (int j = 0; j < 16; ++j) { vals[j] = __expf(vals[j] - mx); sum += vals[j]; }
    sum += __shfl_xor(sum, 1, 32);
    sum += __shfl_xor(sum, 2, 32);
    float inv = 1.f / sum;
    bf16_t* prow = sP + hh * (16 * LDP) + q * LDP + part * 16;
#pragma unroll
    for (int j = 0; j < 16; ++j) prow[j] = (bf16_t)(vals[j] * inv);
  }
  __syncthreads();

  // ---- write AV: mem_upd[q][hh*64+v] ----
  for (int t = wid; t < 16; t += 8) {
    int hh = t >> 2, n0 = t & 3;
    v8f acc = mm_tile<2>(sP + hh * (16 * LDP), LDP,
                         sU + hh * (64 * LDP) + n0 * 16 * LDP, LDP);
    st_tile_bf(sE + hh * 64 + n0 * 16, LDW, acc);
  }
  __syncthreads();

  // ---- M_new = M + mem_upd @ Wwo (fp32->out, bf16->sG) ----
  for (int t = wid; t < 16; t += 8) {
    v8f acc = mm_tile<8>(sE, LDW, WwoT + t * 16 * DD, DD);
    int nn = lane & 15, mb = (lane >> 4) << 3;
    int col = t * 16 + nn;
#pragma unroll
    for (int j = 0; j < 8; ++j) {
      int row = mb + j;
      float vv = acc[j] + Mn[row * DD + col];
      oM[row * DD + col] = vv;
      sG[row * LDW + col] = (bf16_t)vv;
    }
  }
  __syncthreads();

  // ---- read-phase projections ----
  for (int t = wid; t < 16; t += 8) {              // rk -> sD
    v8f acc = mm_tile<8>(sG, LDW, WrkT + t * 16 * DD, DD);
    st_tile_bf(sD + t * 16, LDW, acc);
  }
  for (int t = wid; t < 16; t += 8) {              // rv -> rvT scatter (sP)
    v8f acc = mm_tile<8>(sG, LDW, WrvT + t * 16 * DD, DD);
    int nn = lane & 15, mb = (lane >> 4) << 3;
    int col = t * 16 + nn;
#pragma unroll
    for (int j = 0; j < 8; ++j) {
      int row = mb + j;
      int hh = row >> 2;
      int slot = ((row & 3) << 2) + (col >> 6);
      int v = col & 63;
      sP[hh * (64 * LDR) + v * LDR + slot] = (bf16_t)acc[j];
    }
  }
  for (int t = wid; t < 64; t += 8) {              // rq = h @ Wrq -> sB
    int m0 = t >> 4, n0 = t & 15;
    v8f acc = mm_tile<8>(sH + m0 * 16 * LDW, LDW, WrqT + n0 * 16 * DD, DD);
    st_tile_bf(sB + m0 * 16 * LDW + n0 * 16, LDW, acc);
  }
  for (int e = tid; e < HH_N * 64 * 16; e += 256) { // zero rvT K in [16,32)
    int hh = e >> 10;
    int v  = (e >> 4) & 63;
    int k  = (e & 15) + 16;
    sP[hh * (64 * LDR) + v * LDR + k] = (bf16_t)0.f;
  }
  __syncthreads();

  // ---- read scores: [64q x 16k] per head ----
  for (int t = wid; t < 16; t += 8) {
    int hh = t >> 2, m0 = t & 3;
    v8f acc = {};
#pragma unroll
    for (int ks = 0; ks < 2; ++ks) {
      v16bf a = ld_tile_map(sB, hh * 64 + m0 * 16, ks * 32);
      v16bf b = ld_tile_map(sD, hh * 16, ks * 32);
      acc = __builtin_amdgcn_wmma_f32_16x16x32_bf16(false, a, false, b,
                                                    (short)0, acc, false, false);
    }
    st_tile_f32(sS + hh * (64 * LDSR) + m0 * 16 * LDSR, LDSR, acc);
  }
  __syncthreads();

  // ---- read softmax over 16 slots (one thread per (head,q)) ----
  {
    int hh = tid >> 6, q = tid & 63;
    const float* srow = sS + hh * (64 * LDSR) + q * LDSR;
    float vals[16];
    float mx = -3.4e38f;
#pragma unroll
    for (int j = 0; j < 16; ++j) {
      float s = srow[j] * 0.125f;
      vals[j] = s;
      mx = fmaxf(mx, s);
    }
    float sum = 0.f;
#pragma unroll
    for (int j = 0; j < 16; ++j) { vals[j] = __expf(vals[j] - mx); sum += vals[j]; }
    float inv = 1.f / sum;
    bf16_t* prow = sPr + hh * (64 * LDR) + q * LDR;
#pragma unroll
    for (int j = 0; j < 16; ++j) prow[j] = (bf16_t)(vals[j] * inv);
#pragma unroll
    for (int j = 0; j < 16; ++j) prow[16 + j] = (bf16_t)0.f;
  }
  __syncthreads();

  // ---- read AV: h_upd[q][hh*64+v] (K padded to 32) ----
  for (int t = wid; t < 64; t += 8) {
    int hh = t >> 4, m0 = (t >> 2) & 3, n0 = t & 3;
    v8f acc = mm_tile<1>(sPr + hh * (64 * LDR) + m0 * 16 * LDR, LDR,
                         sP  + hh * (64 * LDR) + n0 * 16 * LDR, LDR);
    st_tile_bf(sU + (m0 * 16) * LDW + hh * 64 + n0 * 16, LDW, acc);
  }
  __syncthreads();

  // ---- h_new = h + h_upd @ Wro -> global ----
  for (int t = wid; t < 64; t += 8) {
    int m0 = t >> 4, n0 = t & 15;
    v8f acc = mm_tile<8>(sU + m0 * 16 * LDW, LDW, WroT + n0 * 16 * DD, DD);
    int nn = lane & 15, mb = (lane >> 4) << 3;
    int col = n0 * 16 + nn;
#pragma unroll
    for (int j = 0; j < 8; ++j) {
      int row = m0 * 16 + mb + j;
      oH[row * DD + col] = acc[j] + hn[row * DD + col];
    }
  }
}

extern "C" void kernel_launch(void* const* d_in, const int* in_sizes, int n_in,
                              void* d_out, int out_size, void* d_ws, size_t ws_size,
                              hipStream_t stream) {
  const float* Mg   = (const float*)d_in[0];
  const float* hg   = (const float*)d_in[1];
  const float* mask = (const float*)d_in[2];

  bf16_t* wsb = (bf16_t*)d_ws;
  // weight order: Wwk Wwv Wwq Wwo Wrk Wrv Wrq Wro (d_in[3..10])
  for (int i = 0; i < 8; ++i)
    wcvt_kernel<<<256, 256, 0, stream>>>((const float*)d_in[3 + i],
                                         wsb + (size_t)i * 65536);

  float* outM = (float*)d_out;
  float* outH = outM + (size_t)NB * KM * DD;

  hipFuncSetAttribute(reinterpret_cast<const void*>(ws_attn_kernel),
                      hipFuncAttributeMaxDynamicSharedMemorySize,
                      (int)SMEM_BYTES);
  ws_attn_kernel<<<NB, 256, SMEM_BYTES, stream>>>(Mg, hg, mask, wsb, outM, outH);
}
